// MultiHeadSelfAttention_57956288692812
// MI455X (gfx1250) — matrix operand order repaired and checked by reading, not compiled, not run on hardware
//
#include <hip/hip_runtime.h>
#include <hip/hip_bf16.h>

// ---------------------------------------------------------------------------
// MHSA for MI455X (gfx1250): bf16 WMMA everywhere, f32 accumulate.
//   B=2, N=2048, D=1024, H=16, Dh=64
//   - GEMMs + flash attention on v_wmma_f32_16x16x32_bf16
//   - A/K tile staging via GLOBAL_LOAD_ASYNC_TO_LDS_B128 (ASYNCcnt)
// ---------------------------------------------------------------------------

typedef __attribute__((ext_vector_type(16))) __bf16 v16bf;
typedef __attribute__((ext_vector_type(8)))  __bf16 v8bf;
typedef __attribute__((ext_vector_type(8)))  float  v8f;

union V16 {
    v16bf v;
    v8bf  h[2];
};

#define WMMA_BF16(a, b, c) \
    __builtin_amdgcn_wmma_f32_16x16x32_bf16(false, (a), false, (b), (short)0, (c), false, false)

// Async 16B global -> LDS copy (per active lane), tracked by ASYNCcnt.
__device__ inline void async_g2l_b128(void* lds_generic, const void* gptr) {
    unsigned lds_off = (unsigned)(size_t)lds_generic;  // LDS aperture: addr[31:0] = LDS offset
    asm volatile("global_load_async_to_lds_b128 %0, %1, off"
                 :: "v"(lds_off), "v"(gptr) : "memory");
}
__device__ inline void wait_asynccnt0() {
    asm volatile("s_wait_asynccnt 0x0" ::: "memory");
}

// ---------------------------------------------------------------------------
// fp32 -> bf16 conversion (vectorized x4; all sizes are multiples of 4)
// ---------------------------------------------------------------------------
__global__ __launch_bounds__(256) void cvt_f32_bf16(const float* __restrict__ in,
                                                    __bf16* __restrict__ out, int n4) {
    int i = blockIdx.x * blockDim.x + threadIdx.x;
    if (i < n4) {
        float4 f = ((const float4*)in)[i];
        __bf16 o[4];
        o[0] = (__bf16)f.x; o[1] = (__bf16)f.y; o[2] = (__bf16)f.z; o[3] = (__bf16)f.w;
        ((ulong1*)out)[i] = *(ulong1*)o;   // 8-byte store of 4 bf16
    }
}

// ---------------------------------------------------------------------------
// Tiled bf16 GEMM: C[M,N] = A[M,K] * B[K,N] + bias[N]
//   block = 256 threads = 8 waves (2 x 4), block tile 64 x 256, K-step 32
//   each wave: 32 x 64 output = 2 x 4 WMMA tiles (8 WMMA / K-step / wave)
// ---------------------------------------------------------------------------
template <bool OUT_BF16>
__global__ __launch_bounds__(256) void gemm_bf16_wmma(const __bf16* __restrict__ A,
                                                      const __bf16* __restrict__ Bm,
                                                      const float* __restrict__ bias,
                                                      void* __restrict__ Cout,
                                                      int M, int N, int K) {
    __shared__ __align__(16) __bf16 As[64][40];    // 64x32 tile, padded
    __shared__ __align__(16) __bf16 Bs[256][48];   // 32x256 tile, TRANSPOSED, padded

    const int t    = threadIdx.x;
    const int lane = t & 31;
    const int w    = t >> 5;
    const int wm   = w >> 2;          // 0..1
    const int wn   = w & 3;           // 0..3
    const int half = lane >> 4;       // 0/1
    const int l16  = lane & 15;

    const int blockM = blockIdx.y * 64;
    const int blockN = blockIdx.x * 256;

    v8f acc[2][4] = {};

    for (int k0 = 0; k0 < K; k0 += 32) {
        // --- stage A: 64 rows x 32 cols, one async 16B copy per thread ---
        {
            int r = t >> 2;
            int c = (t & 3) * 8;
            async_g2l_b128(&As[r][c], A + (size_t)(blockM + r) * K + k0 + c);
        }
        // --- stage B transposed: 32 rows x 256 cols, 32 bf16 per thread ---
        {
            int r = t >> 3;               // 0..31  (k row)
            int c = (t & 7) * 32;         // col base
            const __bf16* src = Bm + (size_t)(k0 + r) * N + blockN + c;
            __bf16 tmp[32];
#pragma unroll
            for (int q = 0; q < 4; ++q)
                *(uint4*)(tmp + 8 * q) = *(const uint4*)(src + 8 * q);
#pragma unroll
            for (int i = 0; i < 32; ++i) Bs[c + i][r] = tmp[i];
        }
        if (k0 + 32 < K) {   // gfx1250 global_prefetch_b8 of next tiles
            __builtin_prefetch(A + (size_t)(blockM + (t >> 2)) * K + k0 + 32, 0, 1);
            __builtin_prefetch(Bm + (size_t)(k0 + 32 + (t >> 3)) * N + blockN + (t & 7) * 32, 0, 1);
        }
        wait_asynccnt0();
        __syncthreads();

        // --- fragments + 8 WMMAs per wave ---
        V16 afr[2], bfr[4];
#pragma unroll
        for (int mb = 0; mb < 2; ++mb) {
            int row = wm * 32 + mb * 16 + l16;
            afr[mb].h[0] = *(const v8bf*)(&As[row][half * 8]);
            afr[mb].h[1] = *(const v8bf*)(&As[row][16 + half * 8]);
        }
#pragma unroll
        for (int nb = 0; nb < 4; ++nb) {
            int col = wn * 64 + nb * 16 + l16;
            bfr[nb].h[0] = *(const v8bf*)(&Bs[col][half * 8]);
            bfr[nb].h[1] = *(const v8bf*)(&Bs[col][16 + half * 8]);
        }
#pragma unroll
        for (int mb = 0; mb < 2; ++mb)
#pragma unroll
            for (int nb = 0; nb < 4; ++nb)
                acc[mb][nb] = WMMA_BF16(afr[mb].v, bfr[nb].v, acc[mb][nb]);

        __syncthreads();
    }

    // --- epilogue: bias add + store ---
#pragma unroll
    for (int mb = 0; mb < 2; ++mb)
#pragma unroll
        for (int nb = 0; nb < 4; ++nb) {
            int col = blockN + wn * 64 + nb * 16 + l16;
            float bval = bias[col];
#pragma unroll
            for (int r = 0; r < 8; ++r) {
                int row = blockM + wm * 32 + mb * 16 + half * 8 + r;
                float v = acc[mb][nb][r] + bval;
                if (OUT_BF16)
                    ((__bf16*)Cout)[(size_t)row * N + col] = (__bf16)v;
                else
                    ((float*)Cout)[(size_t)row * N + col] = v;
            }
        }
}

// ---------------------------------------------------------------------------
// Flash attention, bf16 WMMA, online softmax.
//   grid = (B*H, N/128); block = 256 threads = 8 waves; wave = 16 q rows.
//   qkv layout: [b][n][3072] bf16, q at h*64, k at 1024+h*64, v at 2048+h*64
// ---------------------------------------------------------------------------
__device__ inline float rowmax16(float v) {
#pragma unroll
    for (int off = 8; off > 0; off >>= 1) v = fmaxf(v, __shfl_xor(v, off, 16));
    return v;
}
__device__ inline float rowsum16(float v) {
#pragma unroll
    for (int off = 8; off > 0; off >>= 1) v += __shfl_xor(v, off, 16);
    return v;
}

__global__ __launch_bounds__(256) void attn_bf16_wmma(const __bf16* __restrict__ qkv,
                                                      __bf16* __restrict__ attnb) {
    constexpr int N  = 2048;
    constexpr int D3 = 3072;
    constexpr float SCALE = 0.125f;   // 1/sqrt(64)

    __shared__ __align__(16) __bf16 Ks[32][72];        // K tile row-major
    __shared__ __align__(16) __bf16 Vt[64][40];        // V tile transposed [d][key]
    __shared__ __align__(16) __bf16 Ps[8][16][40];     // per-wave P staging

    const int bh = blockIdx.x;
    const int b  = bh >> 4;
    const int h  = bh & 15;
    const int t    = threadIdx.x;
    const int lane = t & 31;
    const int w    = t >> 5;
    const int half = lane >> 4;
    const int l16  = lane & 15;

    const __bf16* base = qkv + (size_t)b * N * D3;
    const __bf16* Qp = base + h * 64;
    const __bf16* Kp = base + 1024 + h * 64;
    const __bf16* Vp = base + 2048 + h * 64;

    const int qrow0 = blockIdx.y * 128 + w * 16;

    // Q fragments: 16x64 = two k-chunks of 32
    V16 qa[2];
#pragma unroll
    for (int kc = 0; kc < 2; ++kc) {
        const __bf16* qr = Qp + (size_t)(qrow0 + l16) * D3 + kc * 32;
        qa[kc].h[0] = *(const v8bf*)(qr + half * 8);
        qa[kc].h[1] = *(const v8bf*)(qr + 16 + half * 8);
    }

    v8f   O[4] = {};
    float mrow[8], lrow[8];
#pragma unroll
    for (int r = 0; r < 8; ++r) { mrow[r] = -1e30f; lrow[r] = 0.f; }

    for (int kt = 0; kt < N / 32; ++kt) {
        const int key0 = kt * 32;
        // stage K tile (32x64): async 16B copy per thread
        {
            int r = t >> 3, c = (t & 7) * 8;
            async_g2l_b128(&Ks[r][c], Kp + (size_t)(key0 + r) * D3 + c);
        }
        // stage V tile transposed (manual: async op cannot transpose)
        {
            int r = t >> 3, c = (t & 7) * 8;
            __bf16 tmp[8];
            *(uint4*)tmp = *(const uint4*)(Vp + (size_t)(key0 + r) * D3 + c);
#pragma unroll
            for (int i = 0; i < 8; ++i) Vt[c + i][r] = tmp[i];
        }
        wait_asynccnt0();
        __syncthreads();

        // ---- S = Q * K^T : two 16-key sub-tiles, 2 k-chunks each ----
        v8f S[2] = {};
#pragma unroll
        for (int st = 0; st < 2; ++st) {
#pragma unroll
            for (int kc = 0; kc < 2; ++kc) {
                V16 kb;
                const __bf16* kr = &Ks[st * 16 + l16][kc * 32 + half * 8];
                kb.h[0] = *(const v8bf*)(kr);
                kb.h[1] = *(const v8bf*)(kr + 16);
                S[st] = WMMA_BF16(qa[kc].v, kb.v, S[st]);
            }
        }

        // ---- online softmax over 32 new keys ----
        float alpha[8];
#pragma unroll
        for (int r = 0; r < 8; ++r) {
            float s0 = S[0][r] * SCALE;
            float s1 = S[1][r] * SCALE;
            float mt = rowmax16(fmaxf(s0, s1));
            float mn = fmaxf(mrow[r], mt);
            float p0 = __expf(s0 - mn);
            float p1 = __expf(s1 - mn);
            float ps = rowsum16(p0 + p1);
            alpha[r] = __expf(mrow[r] - mn);
            lrow[r]  = lrow[r] * alpha[r] + ps;
            mrow[r]  = mn;
            int rl = half * 8 + r;
            Ps[w][rl][l16]      = (__bf16)p0;
            Ps[w][rl][16 + l16] = (__bf16)p1;
        }

        // ---- O = O*alpha + P * V ----
        V16 pa;
        pa.h[0] = *(const v8bf*)(&Ps[w][l16][half * 8]);
        pa.h[1] = *(const v8bf*)(&Ps[w][l16][16 + half * 8]);
#pragma unroll
        for (int dt = 0; dt < 4; ++dt) {
#pragma unroll
            for (int r = 0; r < 8; ++r) O[dt][r] *= alpha[r];
            V16 vb;
            const __bf16* vr = &Vt[dt * 16 + l16][half * 8];
            vb.h[0] = *(const v8bf*)(vr);
            vb.h[1] = *(const v8bf*)(vr + 16);
            O[dt] = WMMA_BF16(pa.v, vb.v, O[dt]);
        }
        __syncthreads();
    }

    // ---- normalize and write [B,N,D] bf16 ----
#pragma unroll
    for (int dt = 0; dt < 4; ++dt)
#pragma unroll
        for (int r = 0; r < 8; ++r) {
            int row = qrow0 + half * 8 + r;
            int col = h * 64 + dt * 16 + l16;
            attnb[(size_t)b * N * 1024 + (size_t)row * 1024 + col] =
                (__bf16)(O[dt][r] / lrow[r]);
        }
}

// ---------------------------------------------------------------------------
// Host launcher
// ---------------------------------------------------------------------------
extern "C" void kernel_launch(void* const* d_in, const int* in_sizes, int n_in,
                              void* d_out, int out_size, void* d_ws, size_t ws_size,
                              hipStream_t stream) {
    const float* x     = (const float*)d_in[0];   // [2,2048,1024]
    const float* W_qkv = (const float*)d_in[1];   // [1024,3072]
    const float* b_qkv = (const float*)d_in[2];   // [3072]
    const float* W_out = (const float*)d_in[3];   // [1024,1024]
    const float* b_out = (const float*)d_in[4];   // [1024]
    float* out = (float*)d_out;                   // [2,2048,1024]

    const int M = 4096, D = 1024, D3 = 3072;

    char* ws = (char*)d_ws;
    __bf16* xb    = (__bf16*)(ws);                        //  8 MB
    __bf16* wqkvb = (__bf16*)(ws + (8u  << 20));          //  6 MB
    __bf16* woutb = (__bf16*)(ws + (14u << 20));          //  2 MB
    __bf16* qkvb  = (__bf16*)(ws + (16u << 20));          // 24 MB
    __bf16* attnb = (__bf16*)(ws + (40u << 20));          //  8 MB

    // 1) fp32 -> bf16 conversions
    cvt_f32_bf16<<<(M * D / 4 + 255) / 256, 256, 0, stream>>>(x, xb, M * D / 4);
    cvt_f32_bf16<<<(D * D3 / 4 + 255) / 256, 256, 0, stream>>>(W_qkv, wqkvb, D * D3 / 4);
    cvt_f32_bf16<<<(D * D / 4 + 255) / 256, 256, 0, stream>>>(W_out, woutb, D * D / 4);

    // 2) qkv = x @ W_qkv + b_qkv  (bf16 out)   M=4096 N=3072 K=1024
    gemm_bf16_wmma<true><<<dim3(D3 / 256, M / 64), 256, 0, stream>>>(
        xb, wqkvb, b_qkv, qkvb, M, D3, D);

    // 3) flash attention -> attnb [B,N,D] bf16
    attn_bf16_wmma<<<dim3(32, 16), 256, 0, stream>>>(qkvb, attnb);

    // 4) out = attn @ W_out + b_out  (f32 out)  M=4096 N=1024 K=1024
    gemm_bf16_wmma<false><<<dim3(D / 256, M / 64), 256, 0, stream>>>(
        attnb, woutb, b_out, out, M, D, D);
}
